// Encoder_52432960749715
// MI455X (gfx1250) — matrix operand order repaired and checked by reading, not compiled, not run on hardware
//
#include <hip/hip_runtime.h>
#include <stdint.h>

// CDNA5 wave32 WMMA types
typedef __attribute__((ext_vector_type(16))) __bf16 v16bf;
typedef __attribute__((ext_vector_type(8)))  float  v8f;

#define B_   128
#define T_   256
#define D_   1024
#define H_   1024
#define N4   4096      // 4*H
#define EPS  1e-3f

__device__ __forceinline__ float sigm(float x) { return 1.0f / (1.0f + __expf(-x)); }

__device__ __forceinline__ float wred(float v) {
#pragma unroll
    for (int o = 16; o > 0; o >>= 1) v += __shfl_xor(v, o, 32);
    return v;
}

// ---- bf16 hi/lo split (row-major copy): a = hi + lo, fp32-like accuracy in bf16x3 GEMM
__global__ void split_rm(const float* __restrict__ src,
                         __bf16* __restrict__ hi, __bf16* __restrict__ lo, long n) {
    long i = (long)blockIdx.x * blockDim.x + threadIdx.x;
    if (i < n) {
        float f = src[i];
        __bf16 h = (__bf16)f;
        hi[i] = h;
        lo[i] = (__bf16)(f - (float)h);
    }
}

// ---- split + transpose: src [K][N] row-major -> dst [N][K] (K-major, WMMA-B friendly)
__global__ void split_tr(const float* __restrict__ src,
                         __bf16* __restrict__ hiT, __bf16* __restrict__ loT,
                         int K, int N) {
    long i = (long)blockIdx.x * blockDim.x + threadIdx.x;
    long tot = (long)K * N;
    if (i < tot) {
        long k = i / N, c = i % N;
        float f = src[i];
        __bf16 h = (__bf16)f;
        long o = c * (long)K + k;
        hiT[o] = h;
        loT[o] = (__bf16)(f - (float)h);
    }
}

// Build a WMMA A operand (16x32 bf16, row-major source, stride K=1024) for this lane.
// lane = m + 16*sel ; VGPRs 0..3 <- k = sel*8..+7 ; VGPRs 4..7 <- k = 16+sel*8..+7
__device__ __forceinline__ v16bf load_a(const __bf16* base, int k0, int sel) {
    v16bf a;
    ((uint4*)&a)[0] = *(const uint4*)(base + k0 + sel * 8);
    ((uint4*)&a)[1] = *(const uint4*)(base + k0 + 16 + sel * 8);
    return a;
}

// Build a WMMA B operand (32x16 bf16) from K-major storage Bt[n][k], stride K=1024.
// lane = n + 16*kg ; VGPRs 0..3 <- k = kg*8..+7 ; VGPRs 4..7 <- k = 16+kg*8..+7
__device__ __forceinline__ v16bf load_b(const __bf16* bt, int n0, int n, int kg, int k0) {
    const __bf16* p = bt + (size_t)(n0 + n) * 1024 + k0 + kg * 8;
    v16bf b;
    ((uint4*)&b)[0] = *(const uint4*)(p);
    ((uint4*)&b)[1] = *(const uint4*)(p + 16);
    return b;
}

// =====================  Phase 1: zx = LN(x @ W) * gk + bk  =====================
// grid = 2048 blocks (16 rows each), 1024 threads = 32 waves, wave tile = 16x128.
// Raw GEMM tile (16x4096 f32, 256KB) staged in LDS; LN done row-per-wave.
__global__ void gemm_ln_x(const __bf16* __restrict__ xhi, const __bf16* __restrict__ xlo,
                          const __bf16* __restrict__ wThi, const __bf16* __restrict__ wTlo,
                          const float* __restrict__ gk, const float* __restrict__ bk,
                          float* __restrict__ zx) {
    extern __shared__ char smem[];
    float* tile = (float*)smem;                       // 16 x 4096 f32

    const int tid = threadIdx.x, lane = tid & 31, w = tid >> 5;
    const int row0 = blockIdx.x * 16;
    const int m = lane & 15, sel = lane >> 4;
    const int nb = w * 128;

    v8f acc[8];
#pragma unroll
    for (int j = 0; j < 8; ++j) acc[j] = {};

    const __bf16* aph = xhi + (size_t)(row0 + m) * D_;
    const __bf16* apl = xlo + (size_t)(row0 + m) * D_;

    for (int k0 = 0; k0 < D_; k0 += 32) {
        v16bf ah = load_a(aph, k0, sel);
        v16bf al = load_a(apl, k0, sel);
#pragma unroll
        for (int j = 0; j < 8; ++j) {
            const int n0 = nb + j * 16;
            v16bf bh = load_b(wThi, n0, m, sel, k0);
            v16bf bl = load_b(wTlo, n0, m, sel, k0);
            // bf16x3 emulated-fp32 product
            acc[j] = __builtin_amdgcn_wmma_f32_16x16x32_bf16(false, ah, false, bh, (short)0, acc[j], false, false);
            acc[j] = __builtin_amdgcn_wmma_f32_16x16x32_bf16(false, ah, false, bl, (short)0, acc[j], false, false);
            acc[j] = __builtin_amdgcn_wmma_f32_16x16x32_bf16(false, al, false, bh, (short)0, acc[j], false, false);
        }
    }
    // C/D layout: element (m = r + 8*(lane>>4), n = lane&15) in VGPR r
#pragma unroll
    for (int j = 0; j < 8; ++j)
#pragma unroll
        for (int r = 0; r < 8; ++r)
            tile[(r + 8 * sel) * N4 + nb + j * 16 + m] = acc[j][r];
    __syncthreads();

    if (w < 16) {
        const int r = w;
        float s = 0.f, q = 0.f;
        for (int n = lane; n < N4; n += 32) { float v = tile[r * N4 + n]; s += v; q += v * v; }
        s = wred(s); q = wred(q);
        float mu = s * (1.0f / N4);
        float rstd = rsqrtf(q * (1.0f / N4) - mu * mu + EPS);
        float* op = zx + (size_t)(row0 + r) * N4;
        for (int n = lane; n < N4; n += 32)
            op[n] = (tile[r * N4 + n] - mu) * rstd * gk[n] + bk[n];
    }
}

// =====================  Phase 2: persistent recurrence kernel  =====================
// h_new[b] depends only on h[b] => no inter-block sync. 8 blocks x 16 batch rows,
// all 256 timesteps in one launch. h kept in LDS as bf16 hi/lo; U streamed from L2.
// LDS: 16x4096 f32 zh tile (256KB) + 2 x 16x1024 bf16 h (64KB) = 320KB (WGP max).
__global__ void lstm_seq(const float* __restrict__ zx,
                         const __bf16* __restrict__ uThi, const __bf16* __restrict__ uTlo,
                         const float* __restrict__ bias,
                         const float* __restrict__ gr, const float* __restrict__ br,
                         const float* __restrict__ gs, const float* __restrict__ bs,
                         const unsigned char* __restrict__ mask,
                         float* __restrict__ seq,
                         float* __restrict__ hlast, float* __restrict__ clast) {
    extern __shared__ char smem[];
    float*  zh = (float*)smem;                                  // 16 x 4096 f32
    __bf16* hh = (__bf16*)(smem + (size_t)16 * N4 * 4);         // 16 x 1024
    __bf16* hl = (__bf16*)(smem + (size_t)16 * N4 * 4 + (size_t)16 * H_ * 2);

    const int tid = threadIdx.x, lane = tid & 31, w = tid >> 5;
    const int row0 = blockIdx.x * 16;
    const int m = lane & 15, sel = lane >> 4;
    const int nb = w * 128;

    for (int i = tid; i < 16 * H_; i += blockDim.x) { hh[i] = (__bf16)0.0f; hl[i] = (__bf16)0.0f; }
    float creg[32], hreg[32];
#pragma unroll
    for (int i = 0; i < 32; ++i) { creg[i] = 0.f; hreg[i] = 0.f; }
    __syncthreads();

    for (int t = 0; t < T_; ++t) {
        // ---- zh_raw = h @ U  (bf16x3 WMMA, A from LDS, B from L2-resident U^T) ----
        v8f acc[8];
#pragma unroll
        for (int j = 0; j < 8; ++j) acc[j] = {};

        const __bf16* aph = hh + m * H_;
        const __bf16* apl = hl + m * H_;
        for (int k0 = 0; k0 < H_; k0 += 32) {
            v16bf ah = load_a(aph, k0, sel);
            v16bf al = load_a(apl, k0, sel);
#pragma unroll
            for (int j = 0; j < 8; ++j) {
                const int n0 = nb + j * 16;
                v16bf bh = load_b(uThi, n0, m, sel, k0);
                v16bf bl = load_b(uTlo, n0, m, sel, k0);
                acc[j] = __builtin_amdgcn_wmma_f32_16x16x32_bf16(false, ah, false, bh, (short)0, acc[j], false, false);
                acc[j] = __builtin_amdgcn_wmma_f32_16x16x32_bf16(false, ah, false, bl, (short)0, acc[j], false, false);
                acc[j] = __builtin_amdgcn_wmma_f32_16x16x32_bf16(false, al, false, bh, (short)0, acc[j], false, false);
            }
        }
#pragma unroll
        for (int j = 0; j < 8; ++j)
#pragma unroll
            for (int r = 0; r < 8; ++r)
                zh[(r + 8 * sel) * N4 + nb + j * 16 + m] = acc[j][r];
        __syncthreads();

        // ---- LN(zh) + gates + state LN, one batch row per wave (waves 0..15) ----
        if (w < 16) {
            const int r = w;
            const int b = row0 + r;
            float s = 0.f, q = 0.f;
            for (int n = lane; n < N4; n += 32) { float v = zh[r * N4 + n]; s += v; q += v * v; }
            s = wred(s); q = wred(q);
            float mu = s * (1.0f / N4);
            float rstd = rsqrtf(q * (1.0f / N4) - mu * mu + EPS);

            const float* zxr = zx + ((size_t)b * T_ + t) * N4;
            float cn_s[32];
            float cs = 0.f, cq = 0.f;
#pragma unroll
            for (int it = 0; it < 32; ++it) {
                int j = lane + it * 32;
                float zi = (zh[r * N4 + j       ] - mu) * rstd * gr[j       ] + br[j       ] + zxr[j       ] + bias[j       ];
                float zf = (zh[r * N4 + j + 1024] - mu) * rstd * gr[j + 1024] + br[j + 1024] + zxr[j + 1024] + bias[j + 1024];
                float zg = (zh[r * N4 + j + 2048] - mu) * rstd * gr[j + 2048] + br[j + 2048] + zxr[j + 2048] + bias[j + 2048];
                float cn = sigm(zf) * creg[it] + sigm(zi) * tanhf(zg);
                cn_s[it] = cn; cs += cn; cq += cn * cn;
            }
            cs = wred(cs); cq = wred(cq);
            float muc = cs * (1.0f / H_);
            float rstdc = rsqrtf(cq * (1.0f / H_) - muc * muc + EPS);

            const bool mk = mask[(size_t)b * T_ + t] != 0;
            float* sp = seq + ((size_t)b * T_ + t) * H_;
#pragma unroll
            for (int it = 0; it < 32; ++it) {
                int j = lane + it * 32;
                float zo = (zh[r * N4 + j + 3072] - mu) * rstd * gr[j + 3072] + br[j + 3072] + zxr[j + 3072] + bias[j + 3072];
                float cl = (cn_s[it] - muc) * rstdc * gs[j] + bs[j];
                float hn = sigm(zo) * tanhf(cl);
                float ho = mk ? hn : hreg[it];
                float co = mk ? cl : creg[it];
                hreg[it] = ho; creg[it] = co;
                sp[j] = ho;
                __bf16 hb = (__bf16)ho;
                hh[r * H_ + j] = hb;
                hl[r * H_ + j] = (__bf16)(ho - (float)hb);
                if (t == T_ - 1) {
                    hlast[(size_t)b * H_ + j] = ho;
                    clast[(size_t)b * H_ + j] = co;
                }
            }
        }
        __syncthreads();
    }
}

extern "C" void kernel_launch(void* const* d_in, const int* in_sizes, int n_in,
                              void* d_out, int out_size, void* d_ws, size_t ws_size,
                              hipStream_t stream) {
    (void)in_sizes; (void)n_in; (void)out_size; (void)ws_size;
    const float*         x    = (const float*)d_in[0];
    const unsigned char* mask = (const unsigned char*)d_in[1];
    const float*         W    = (const float*)d_in[2];
    const float*         U    = (const float*)d_in[3];
    const float*         bias = (const float*)d_in[4];
    const float*         gk   = (const float*)d_in[5];
    const float*         bk   = (const float*)d_in[6];
    const float*         gr   = (const float*)d_in[7];
    const float*         br   = (const float*)d_in[8];
    const float*         gs   = (const float*)d_in[9];
    const float*         bs   = (const float*)d_in[10];

    // workspace carve-out
    char* ws = (char*)d_ws;
    size_t o = 0;
    __bf16* xhi  = (__bf16*)(ws + o); o += (size_t)B_ * T_ * D_ * 2;   // 64 MB
    __bf16* xlo  = (__bf16*)(ws + o); o += (size_t)B_ * T_ * D_ * 2;   // 64 MB
    __bf16* wThi = (__bf16*)(ws + o); o += (size_t)N4 * D_ * 2;        // 8 MB
    __bf16* wTlo = (__bf16*)(ws + o); o += (size_t)N4 * D_ * 2;
    __bf16* uThi = (__bf16*)(ws + o); o += (size_t)N4 * H_ * 2;
    __bf16* uTlo = (__bf16*)(ws + o); o += (size_t)N4 * H_ * 2;
    float*  zx   = (float*)(ws + o);  o += (size_t)B_ * T_ * N4 * 4;   // 512 MB

    // phase 0: precision-split conversions
    long nx = (long)B_ * T_ * D_;
    split_rm<<<(unsigned)((nx + 255) / 256), 256, 0, stream>>>(x, xhi, xlo, nx);
    long nw = (long)D_ * N4;
    split_tr<<<(unsigned)((nw + 255) / 256), 256, 0, stream>>>(W, wThi, wTlo, D_, N4);
    split_tr<<<(unsigned)((nw + 255) / 256), 256, 0, stream>>>(U, uThi, uTlo, H_, N4);

    // phase 1: zx = LN(x @ W)  (2048 blocks, 256KB dynamic LDS each)
    gemm_ln_x<<<(B_ * T_) / 16, 1024, 16 * N4 * 4, stream>>>(xhi, xlo, wThi, wTlo, gk, bk, zx);

    // phase 2: persistent recurrence (8 blocks, 320KB dynamic LDS each)
    float* seq   = (float*)d_out;
    float* hlast = seq + (size_t)B_ * T_ * H_;
    float* clast = hlast + (size_t)B_ * H_;
    lstm_seq<<<B_ / 16, 1024, 16 * N4 * 4 + 2 * 16 * H_ * 2, stream>>>(
        zx, uThi, uTlo, bias, gr, br, gs, bs, mask, seq, hlast, clast);
}
